// ODConvTranspose1D_10703058502155
// MI455X (gfx1250) — matrix operand order repaired and checked by reading, not compiled, not run on hardware
//
#include <hip/hip_runtime.h>
#include <math.h>

typedef unsigned short u16_t;
typedef __attribute__((ext_vector_type(16))) __bf16 v16bf;
typedef __attribute__((ext_vector_type(8)))  float  v8f;

struct __align__(16) U128 { unsigned int x, y, z, w; };
union Frag { U128 q[2]; v16bf v; };

#define B_     8
#define L_     2048
#define CIN_   512
#define COUT_  256
#define K_     16
#define RED_   128
#define LOUT_  16384
#define XROWS  2050   // columns m = -1 .. 2048 (zero padded ends)
#define LDP    40     // LDS row pitch in halves (16B aligned, bank-spread)

__device__ __forceinline__ u16_t f2bf(float f) {
  unsigned int u = __float_as_uint(f);
  u += 0x7FFFu + ((u >> 16) & 1u);     // round to nearest even
  return (u16_t)(u >> 16);
}

// ---------------- pooled mean: p[b,c] = mean_m x[b,c,m] ----------------
__global__ __launch_bounds__(256) void pool_k(const float* __restrict__ x,
                                              float* __restrict__ pool) {
  __shared__ float red[256];
  const int c = blockIdx.x, b = blockIdx.y, tid = threadIdx.x;
  const float* row = x + ((size_t)(b * CIN_ + c)) * L_;
  float s = 0.f;
  for (int m = tid; m < L_; m += 256) s += row[m];
  red[tid] = s; __syncthreads();
  for (int st = 128; st > 0; st >>= 1) {
    if (tid < st) red[tid] += red[tid + st];
    __syncthreads();
  }
  if (tid == 0) pool[b * CIN_ + c] = red[0] * (1.f / (float)L_);
}

// ---------------- four attention heads (tiny dense MLPs) ----------------
struct AttnParams {
  const float *w1[4], *b1[4], *w2[4], *b2[4];
  float* outp[4];
  int od[4];
};

__global__ __launch_bounds__(128) void attn_k(const float* __restrict__ pool, AttnParams P) {
  __shared__ float sp[CIN_];
  __shared__ float sh[RED_];
  const int b = blockIdx.x, tid = threadIdx.x;
  for (int c = tid; c < CIN_; c += 128) sp[c] = pool[b * CIN_ + c];
  __syncthreads();
  for (int hd = 0; hd < 4; ++hd) {
    const float* w1 = P.w1[hd];
    float a = P.b1[hd][tid];
    for (int c = 0; c < CIN_; ++c) a += w1[tid * CIN_ + c] * sp[c];
    sh[tid] = a > 0.f ? a : 0.f;
    __syncthreads();
    const int od = P.od[hd];
    const float* w2 = P.w2[hd];
    for (int o = tid; o < od; o += 128) {
      float acc = P.b2[hd][o];
      for (int r = 0; r < RED_; ++r) acc += w2[o * RED_ + r] * sh[r];
      P.outp[hd][b * od + o] = 1.f / (1.f + expf(-acc));
    }
    __syncthreads();
  }
}

// -------- Xs[b][m+1][c] = bf16(ia[b,c] * x[b,c,m]) (LDS-tiled transpose) --------
__global__ __launch_bounds__(256) void prep_x_k(const float* __restrict__ x,
                                                const float* __restrict__ ia,
                                                u16_t* __restrict__ Xs) {
  __shared__ float t[64][65];
  const int m0 = blockIdx.x * 64, c0 = blockIdx.y * 64, b = blockIdx.z, tid = threadIdx.x;
  const int col = tid & 63, rowq = tid >> 6;
#pragma unroll
  for (int i = 0; i < 16; ++i) {
    int cl = rowq + i * 4;
    t[cl][col] = x[((size_t)(b * CIN_ + c0 + cl)) * L_ + m0 + col];
  }
  __syncthreads();
#pragma unroll
  for (int i = 0; i < 16; ++i) {
    int ml = rowq + i * 4;
    int cl = col;
    float v = t[cl][ml] * ia[b * CIN_ + c0 + cl];
    Xs[(((size_t)(b * XROWS + m0 + ml + 1)) << 9) + (c0 + cl)] = f2bf(v);
  }
}

__global__ void pad_k(u16_t* __restrict__ Xs) {
  const int b = blockIdx.x, tid = threadIdx.x;
  for (int c = tid; c < CIN_; c += 256) {
    Xs[((size_t)(b * XROWS + 0)) * CIN_ + c] = 0;
    Xs[((size_t)(b * XROWS + 2049)) * CIN_ + c] = 0;
  }
}

// -------- Wt[k][o][c] = bf16(w[c][o][k]) (LDS-tiled transpose) --------
__global__ __launch_bounds__(256) void prep_w_k(const float* __restrict__ w,
                                                u16_t* __restrict__ Wt) {
  __shared__ float t[64][65];
  const int ok0 = blockIdx.x * 64, c0 = blockIdx.y * 64, tid = threadIdx.x;
  const int col = tid & 63, rowq = tid >> 6;
#pragma unroll
  for (int i = 0; i < 16; ++i) {
    int cl = rowq + i * 4;
    t[cl][col] = w[((size_t)(c0 + cl)) * (COUT_ * K_) + ok0 + col];
  }
  __syncthreads();
#pragma unroll
  for (int i = 0; i < 16; ++i) {
    int okl = rowq + i * 4;
    int cl = col;
    int ok = ok0 + okl;
    int k = ok & 15, o = ok >> 4;
    Wt[((size_t)(k * COUT_ + o)) * CIN_ + c0 + cl] = f2bf(t[cl][okl]);
  }
}

// ---------------- main WMMA GEMM: per (b, phase p) ----------------
// C0[o,m] = sum_c Wt[p][o][c]   * X'[m  ][c]
// C1[o,m] = sum_c Wt[p+8][o][c] * X'[m-1][c]
// out[b,o,8m+p-4] = sa*oa[o]*(ka[p]*C0 + ka[p+8]*C1) + bias[o]
// Staging uses CDNA5 async global->LDS loads (ASYNCcnt) with LDS double buffering.
__global__ __launch_bounds__(256) void gemm_k(
    const u16_t* __restrict__ Wt, const u16_t* __restrict__ Xs,
    const float* __restrict__ ka, const float* __restrict__ sa,
    const float* __restrict__ oa, const float* __restrict__ bias,
    float* __restrict__ out) {
  __shared__ u16_t lsA[2][2 * 128 * LDP];  // [buf][tapset][o=128][cc=32] padded pitch
  __shared__ u16_t lsB[2][65 * LDP];       // [buf][m rows m_base-1..+63][cc=32]

  const int tid = threadIdx.x;
  const int b = blockIdx.z >> 3;
  const int p = blockIdx.z & 7;
  const int o_blk = blockIdx.y * 128;
  const int m_base = blockIdx.x * 64;

  const int lane = tid & 31;
  const int wave = tid >> 5;
  const int wo = (wave & 3) * 32;        // wave o-offset within tile
  const int wm = (wave >> 2) * 32;       // wave m-offset within tile
  const int lhalf = lane >> 4;           // 0: lanes 0-15, 1: lanes 16-31
  const int l15 = lane & 15;

  // Issue one K-slice of async global->LDS loads: exactly 6 async instrs per wave.
  auto issue_tile = [&](int ksu, u16_t* bufA, u16_t* bufB) {
    const int cb = ksu * 32;
    // A (both taps): 2 * 128 rows * 32 halves = 1024 x 16B chunks (4 per thread)
#pragma unroll
    for (int i = 0; i < 4; ++i) {
      int idx = tid + i * 256;
      int s = idx >> 9;
      int r = (idx >> 2) & 127;
      int q = idx & 3;
      int ksel = p + s * 8;
      const u16_t* g = Wt + (((size_t)(ksel * COUT_ + o_blk + r)) << 9) + cb + q * 8;
      unsigned loff = (unsigned)(uintptr_t)(bufA + (s * 128 + r) * LDP + q * 8);
      asm volatile("global_load_async_to_lds_b128 %0, %1, off"
                   :: "v"(loff), "v"(g) : "memory");
    }
    // B: 65 rows * 32 halves = 260 x 16B chunks; second pass loads the 4 tail
    // chunks redundantly from every thread so the async count stays fixed.
#pragma unroll
    for (int i = 0; i < 2; ++i) {
      int idx = (i == 0) ? tid : (256 + (tid & 3));
      int j = idx >> 2;
      int q = idx & 3;
      int row = m_base + j; if (row > 2049) row = 2049;
      const u16_t* g = Xs + (((size_t)(b * XROWS + row)) << 9) + cb + q * 8;
      unsigned loff = (unsigned)(uintptr_t)(bufB + j * LDP + q * 8);
      asm volatile("global_load_async_to_lds_b128 %0, %1, off"
                   :: "v"(loff), "v"(g) : "memory");
    }
  };

  const v8f vzero = {0.f, 0.f, 0.f, 0.f, 0.f, 0.f, 0.f, 0.f};
  v8f acc[2][2][2];                      // [tapset][o-tile][m-tile]
#pragma unroll
  for (int s = 0; s < 2; ++s)
#pragma unroll
    for (int i = 0; i < 2; ++i)
#pragma unroll
      for (int j = 0; j < 2; ++j) acc[s][i][j] = vzero;

  issue_tile(0, lsA[0], lsB[0]);

  for (int ksu = 0; ksu < 16; ++ksu) {
    if (ksu < 15) {
      issue_tile(ksu + 1, lsA[(ksu + 1) & 1], lsB[(ksu + 1) & 1]);
      // 6 outstanding = the batch just issued; waits for the current tile only.
      asm volatile("s_wait_asynccnt 6" ::: "memory");
    } else {
      asm volatile("s_wait_asynccnt 0" ::: "memory");
    }
    __syncthreads();   // current tile visible to all waves

    const u16_t* curA = lsA[ksu & 1];
    const u16_t* curB = lsB[ksu & 1];

    // A frag (16x32 bf16): lane l: M=l&15, halves [K0..K0+7],[K0+16..K0+23], K0=8*(l>=16)
    Frag afr[2][2];
#pragma unroll
    for (int s = 0; s < 2; ++s)
#pragma unroll
      for (int to = 0; to < 2; ++to) {
        const u16_t* base = curA + (s * 128 + wo + to * 16 + l15) * LDP + 8 * lhalf;
        afr[s][to].q[0] = *(const U128*)(base);
        afr[s][to].q[1] = *(const U128*)(base + 16);
      }
    // B frag (32x16 bf16): lane l: N=l&15, halves [K0..K0+7],[K0+16..K0+23], K0=8*(l>=16)
    // tap0 reads LDS row (m_local+1) -> column m ; tap1 reads row m_local -> column m-1
    Frag bfr[2][2];
#pragma unroll
    for (int s = 0; s < 2; ++s)
#pragma unroll
      for (int tm = 0; tm < 2; ++tm) {
        const u16_t* base = curB + (wm + tm * 16 + l15 + (1 - s)) * LDP + 8 * lhalf;
        bfr[s][tm].q[0] = *(const U128*)(base);
        bfr[s][tm].q[1] = *(const U128*)(base + 16);
      }
#pragma unroll
    for (int s = 0; s < 2; ++s)
#pragma unroll
      for (int to = 0; to < 2; ++to)
#pragma unroll
        for (int tm = 0; tm < 2; ++tm)
          acc[s][to][tm] = __builtin_amdgcn_wmma_f32_16x16x32_bf16(
              false, afr[s][to].v, false, bfr[s][tm].v,
              (short)0, acc[s][to][tm], false, false);
    __syncthreads();   // done reading this buffer before it is overwritten
  }

  // epilogue: combine taps with ka, scale by sa*oa, add bias, scatter to t = 8m+p-4
  const float sab = sa[b];
  const float kap  = ka[b * K_ + p];
  const float kap8 = ka[b * K_ + p + 8];
#pragma unroll
  for (int to = 0; to < 2; ++to) {
    const int o0 = o_blk + wo + to * 16 + 8 * lhalf;   // D vgpr r -> o = o0 + r
    float sc[8], bs[8];
#pragma unroll
    for (int r = 0; r < 8; ++r) {
      sc[r] = sab * oa[b * COUT_ + o0 + r];
      bs[r] = bias[o0 + r];
    }
#pragma unroll
    for (int tm = 0; tm < 2; ++tm) {
      const int m_g = m_base + wm + tm * 16 + l15;
      const int t = 8 * m_g + p - 4;
      if (m_g <= 2048 && t >= 0 && t < LOUT_) {
#pragma unroll
        for (int r = 0; r < 8; ++r) {
          float v = kap * acc[0][to][tm][r] + kap8 * acc[1][to][tm][r];
          out[(((size_t)(b * COUT_ + o0 + r)) << 14) + t] = sc[r] * v + bs[r];
        }
      }
    }
  }
}

// ---------------- host launch ----------------
extern "C" void kernel_launch(void* const* d_in, const int* in_sizes, int n_in,
                              void* d_out, int out_size, void* d_ws, size_t ws_size,
                              hipStream_t stream) {
  (void)in_sizes; (void)n_in; (void)out_size; (void)ws_size;
  const float* x    = (const float*)d_in[0];
  const float* w    = (const float*)d_in[1];
  const float* bias = (const float*)d_in[2];

  char* ws = (char*)d_ws;
  size_t off = 0;
  auto take = [&](size_t bytes) {
    char* ptr = ws + off;
    off = (off + bytes + 255) & ~(size_t)255;
    return ptr;
  };

  u16_t* Xs   = (u16_t*)take((size_t)B_ * XROWS * CIN_ * 2);   // 16.8 MB
  u16_t* Wt   = (u16_t*)take((size_t)K_ * COUT_ * CIN_ * 2);   //  4.2 MB
  float* pool = (float*)take((size_t)B_ * CIN_ * 4);
  float* kaA  = (float*)take((size_t)B_ * K_ * 4);
  float* saA  = (float*)take((size_t)B_ * 4);
  float* iaA  = (float*)take((size_t)B_ * CIN_ * 4);
  float* oaA  = (float*)take((size_t)B_ * COUT_ * 4);

  float* out = (float*)d_out;

  pool_k<<<dim3(CIN_, B_), 256, 0, stream>>>(x, pool);

  AttnParams P;
  const int base[4] = {3, 7, 11, 15};
  float* outs[4] = {kaA, saA, iaA, oaA};
  const int ods[4] = {K_, 1, CIN_, COUT_};
  for (int h = 0; h < 4; ++h) {
    P.w1[h] = (const float*)d_in[base[h] + 0];
    P.b1[h] = (const float*)d_in[base[h] + 1];
    P.w2[h] = (const float*)d_in[base[h] + 2];
    P.b2[h] = (const float*)d_in[base[h] + 3];
    P.outp[h] = outs[h];
    P.od[h] = ods[h];
  }
  attn_k<<<dim3(B_), 128, 0, stream>>>(pool, P);

  prep_x_k<<<dim3(L_ / 64, CIN_ / 64, B_), 256, 0, stream>>>(x, iaA, Xs);
  pad_k<<<dim3(B_), 256, 0, stream>>>(Xs);
  prep_w_k<<<dim3((COUT_ * K_) / 64, CIN_ / 64), 256, 0, stream>>>(w, Wt);

  // grid: 33 m-blocks (covers m = 0..2048), 2 o-blocks, 8 batches x 8 phases
  gemm_k<<<dim3(33, 2, B_ * 8), 256, 0, stream>>>(Wt, Xs, kaA, saA, oaA, bias, out);
}
